// ProteinGVPEncoder_78065325572150
// MI455X (gfx1250) — matrix operand order repaired and checked by reading, not compiled, not run on hardware
//
#include <hip/hip_runtime.h>
#include <math.h>

#define NN 8192
#define NE 131072
#define SH 256
#define NRES 1024

typedef __attribute__((ext_vector_type(16))) _Float16 v16h;
typedef __attribute__((ext_vector_type(8)))  _Float16 v8h;
typedef __attribute__((ext_vector_type(8)))  float    v8f;

__device__ __forceinline__ float sigm(float x) { return 1.f / (1.f + __expf(-x)); }

// ---------------------------------------------------------------------------
// WMMA micro-GEMM: D(16x16,f32) += A(16xK,f16 in LDS, row stride lda) *
// B(KxO f16, row-major [k][o], col block n0). K = 32*ksteps.
// A fragment layout (ISA 7.12.2, 16-bit A 16x32): lane m = lane&15,
// elements 0..7 = K kb..kb+7, 8..15 = K 16+kb..16+kb+7, kb = lane<16?0:8.
// B fragment: lane n = lane&15, elements in same K order as A (column N n0+n).
// D: lane n = lane&15, vgpr r -> M = r + (lane<16?0:8).
// ---------------------------------------------------------------------------
__device__ __forceinline__ v8f wmma_gemm(const _Float16* A, int lda,
                                         const _Float16* B, int ldb,
                                         int n0, int ksteps) {
  v8f acc = {0.f, 0.f, 0.f, 0.f, 0.f, 0.f, 0.f, 0.f};
  const int lane = threadIdx.x & 31;
  const int kb = (lane < 16) ? 0 : 8;
  const _Float16* arow = A + (lane & 15) * lda;
  for (int t = 0; t < ksteps; ++t) {
    const int k0 = t * 32;
    v8h a0 = *(const v8h*)(arow + k0 + kb);
    v8h a1 = *(const v8h*)(arow + k0 + kb + 16);
    v16h av = __builtin_shufflevector(a0, a1, 0, 1, 2, 3, 4, 5, 6, 7,
                                      8, 9, 10, 11, 12, 13, 14, 15);
    v16h bv;
    for (int s = 0; s < 16; ++s)
      bv[s] = B[(k0 + kb + (s < 8 ? s : s + 8)) * ldb + n0 + (lane & 15)];
    acc = __builtin_amdgcn_wmma_f32_16x16x32_f16(false, av, false, bv,
                                                 (short)0, acc, false, false);
  }
  return acc;
}

// ------------------------------ utility kernels ----------------------------
__global__ void k_zero(float* p, int n) {
  int i = blockIdx.x * 256 + threadIdx.x;
  if (i < n) p[i] = 0.f;
}

// W [O][K] f32 -> Bt [Kpad][O] f16 (zero-pad rows K..Kpad)
__global__ void k_prep(const float* W, _Float16* Bt, int O, int K, int Kpad) {
  int i = blockIdx.x * 256 + threadIdx.x;
  if (i >= Kpad * O) return;
  int k = i / O, o = i % O;
  Bt[i] = (k < K) ? (_Float16)W[o * K + k] : (_Float16)0.f;
}

__global__ void k_deg(const int* EI, float* DEG) {
  int e = blockIdx.x * 256 + threadIdx.x;
  if (e < NE) atomicAdd(&DEG[EI[NE + e]], 1.f);
}

__global__ void k_rcnt(const int* R, float* C) {
  int i = blockIdx.x * 256 + threadIdx.x;
  if (i < NN) atomicAdd(&C[R[i]], 1.f);
}

__global__ void k_div(float* OUT, const float* C) {
  int i = blockIdx.x * 256 + threadIdx.x;
  if (i < NRES * 128) OUT[i] /= fmaxf(C[i >> 7], 1.f);
}

// ------------------------- encoder + input_gvp -----------------------------
// v0 == 0 in the reference, so input_gvp's vh == 0, v_out == 0, and the norm
// features are all sqrt(EPS) = 1e-4.
__global__ __launch_bounds__(64) void k_encode(
    const float* X, const float* w1, const float* b1, const float* lg,
    const float* lb, const float* w2, const float* b2, const float* ws_w,
    const float* ws_b, float* S, float* Vv) {
  __shared__ float xr[17], h[34], h2[34], s0[17];
  int n = blockIdx.x, t = threadIdx.x;
  if (t < 17) xr[t] = X[n * 17 + t];
  __syncthreads();
  if (t < 34) {
    float a = b1[t];
    for (int k = 0; k < 17; ++k) a += w1[t * 17 + k] * xr[k];
    h[t] = a;
  }
  __syncthreads();
  if (t < 34) {
    float mu = 0.f;
    for (int j = 0; j < 34; ++j) mu += h[j];
    mu *= (1.f / 34.f);
    float var = 0.f;
    for (int j = 0; j < 34; ++j) { float d = h[j] - mu; var += d * d; }
    var *= (1.f / 34.f);
    h2[t] = fmaxf((h[t] - mu) * rsqrtf(var + 1e-5f) * lg[t] + lb[t], 0.f);
  }
  __syncthreads();
  if (t < 17) {
    float a = b2[t];
    for (int j = 0; j < 34; ++j) a += w2[t * 34 + j] * h2[j];
    s0[t] = a;
  }
  __syncthreads();
  for (int o = t; o < SH; o += 64) {
    float a = ws_b[o];
    const float* wr = ws_w + o * 49;
    for (int k = 0; k < 17; ++k) a += wr[k] * s0[k];
    for (int k = 17; k < 49; ++k) a += wr[k] * 1e-4f;  // sqrt(EPS)
    S[n * SH + o] = fmaxf(a, 0.f);
  }
  for (int q = t; q < 96; q += 64) Vv[n * 96 + q] = 0.f;
}

// ------------------------------- edge geometry -----------------------------
__global__ void k_geom(const int* EI, const float* POS, float* U, float* DIST,
                       float* ES) {
  int e = blockIdx.x * 256 + threadIdx.x;
  if (e >= NE) return;
  int s = EI[e], d = EI[NE + e];
  float vx = POS[s * 3 + 0] - POS[d * 3 + 0];
  float vy = POS[s * 3 + 1] - POS[d * 3 + 1];
  float vz = POS[s * 3 + 2] - POS[d * 3 + 2];
  float dn = sqrtf(fmaxf(vx * vx + vy * vy + vz * vz, 1e-8f));
  float inv = 1.f / dn;
  U[e * 3 + 0] = vx * inv;
  U[e * 3 + 1] = vy * inv;
  U[e * 3 + 2] = vz * inv;
  for (int j = 0; j < 16; ++j) {
    float mu = 8.f * (float)j / 15.f;
    float z = (dn - mu) * 2.f;  // sigma = 0.5
    float f = __expf(-z * z);
    DIST[e * 16 + j] = f;
    ES[e * 16 + j] = f;
  }
}

// ---------------------- fused edge-message kernel (WMMA) -------------------
// one block = 16 edges; msg0 (K=608 pad of 593) -> gate -> msg1 (K=288) ->
// gate -> atomic scatter-add into node accumulators.
__global__ __launch_bounds__(256) void k_msg(
    const int* EI, const float* S, const float* Vv, const float* ES,
    const float* U, const float* wh0, const float* wsb0, const float* wv0,
    const float* wsvb0, const float* wh1, const float* wsb1, const float* wv1,
    const float* wsvb1, const _Float16* B0, const _Float16* Bg0,
    const _Float16* B1, const _Float16* Bg1, float* ACC_S, float* ACC_V) {
  extern __shared__ char smem_raw[];
  _Float16* A0 = (_Float16*)smem_raw;     // 16x608
  _Float16* A1 = A0 + 16 * 608;           // 16x288
  _Float16* GA = A1 + 16 * 288;           // 16x256
  float* mvF = (float*)(GA + 16 * 256);   // 16x3x65
  float* vhF = mvF + 3120;                // 16x3x65
  float* vtF = vhF + 3120;                // 16x3x32
  float* mv1F = vtF + 1536;               // 16x3x32
  float* vh1F = mv1F + 1536;              // 16x3x32
  float* g0F = vh1F + 1536;               // 16x32
  float* g1F = g0F + 512;                 // 16x32
  float* so1 = g1F + 512;                 // 16x256 f32
  int* srcI = (int*)(so1 + 4096);
  int* dstI = srcI + 16;

  const int tid = threadIdx.x, wave = tid >> 5, lane = tid & 31;
  const int e0 = blockIdx.x * 16;
  if (tid < 16) { srcI[tid] = EI[e0 + tid]; dstI[tid] = EI[NE + e0 + tid]; }
  __syncthreads();

  // gather scalar features [s_src | es | s_dst] -> f16
  for (int i = tid; i < 16 * 528; i += 256) {
    int e = i / 528, k = i % 528;
    float val;
    if (k < 256) val = S[srcI[e] * SH + k];
    else if (k < 272) val = ES[(e0 + e) * 16 + (k - 256)];
    else val = S[dstI[e] * SH + (k - 272)];
    A0[e * 608 + k] = (_Float16)val;
  }
  // gather mv [e][c][ch]  ch: 0..31 src, 32 = u, 33..64 dst
  for (int i = tid; i < 16 * 195; i += 256) {
    int e = i / 195, r = i % 195, c = r / 65, ch = r % 65;
    float val;
    if (ch < 32) val = Vv[srcI[e] * 96 + ch * 3 + c];
    else if (ch == 32) val = U[(e0 + e) * 3 + c];
    else val = Vv[dstI[e] * 96 + (ch - 33) * 3 + c];
    mvF[(e * 3 + c) * 65 + ch] = val;
  }
  __syncthreads();
  // vh = wh0 @ mv  (65x65 per component)
  for (int i = tid; i < 16 * 195; i += 256) {
    int e = i / 195, r = i % 195, c = r / 65, h = r % 65;
    const float* wr = wh0 + h * 65;
    const float* mr = mvF + (e * 3 + c) * 65;
    float a = 0.f;
    for (int j = 0; j < 65; ++j) a += wr[j] * mr[j];
    vhF[(e * 3 + c) * 65 + h] = a;
  }
  __syncthreads();
  // ||vh|| -> A0[528..592], zero pad 593..607
  for (int i = tid; i < 16 * 80; i += 256) {
    int e = i / 80, h = i % 80;
    _Float16 o;
    if (h < 65) {
      float x = vhF[(e * 3 + 0) * 65 + h], y = vhF[(e * 3 + 1) * 65 + h],
            z = vhF[(e * 3 + 2) * 65 + h];
      o = (_Float16)sqrtf(fmaxf(x * x + y * y + z * z, 1e-8f));
    } else o = (_Float16)0.f;
    A0[e * 608 + 528 + h] = o;
  }
  __syncthreads();
  // msg0 s_out GEMM: [16,608] x [608,256]
  {
    const int n0 = wave * 32;
    v8f ac0 = wmma_gemm(A0, 608, B0, 256, n0, 19);
    v8f ac1 = wmma_gemm(A0, 608, B0, 256, n0 + 16, 19);
    const int nn = lane & 15, mB = (lane < 16) ? 0 : 8;
    for (int r = 0; r < 8; ++r) {
      int m = mB + r;
      float x0 = ac0[r] + wsb0[n0 + nn];
      GA[m * 256 + n0 + nn] = (_Float16)sigm(x0);       // gate input (act)
      A1[m * 288 + n0 + nn] = (_Float16)fmaxf(x0, 0.f); // relu -> msg1 input
      float x1 = ac1[r] + wsb0[n0 + 16 + nn];
      GA[m * 256 + n0 + 16 + nn] = (_Float16)sigm(x1);
      A1[m * 288 + n0 + 16 + nn] = (_Float16)fmaxf(x1, 0.f);
    }
  }
  __syncthreads();
  // gate0 GEMM (waves 0..1) : [16,256] x [256,32]; others do wv0 einsum
  if (wave < 2) {
    const int n0 = wave * 16;
    v8f g = wmma_gemm(GA, 256, Bg0, 32, n0, 8);
    const int nn = lane & 15, mB = (lane < 16) ? 0 : 8;
    for (int r = 0; r < 8; ++r)
      g0F[(mB + r) * 32 + n0 + nn] = sigm(g[r] + wsvb0[n0 + nn]);
  }
  for (int i = tid; i < 16 * 96; i += 256) {
    int e = i / 96, r = i % 96, c = r / 32, o = r % 32;
    const float* wr = wv0 + o * 65;
    const float* vr = vhF + (e * 3 + c) * 65;
    float a = 0.f;
    for (int j = 0; j < 65; ++j) a += wr[j] * vr[j];
    vtF[(e * 3 + c) * 32 + o] = a;
  }
  __syncthreads();
  for (int i = tid; i < 16 * 96; i += 256) {  // mv1 = v_out0 * gate0
    int e = i / 96, r = i % 96, c = r / 32, o = r % 32;
    mv1F[(e * 3 + c) * 32 + o] = vtF[(e * 3 + c) * 32 + o] * g0F[e * 32 + o];
  }
  __syncthreads();
  for (int i = tid; i < 16 * 96; i += 256) {  // vh1 = wh1 @ mv1
    int e = i / 96, r = i % 96, c = r / 32, h = r % 32;
    const float* wr = wh1 + h * 32;
    const float* mr = mv1F + (e * 3 + c) * 32;
    float a = 0.f;
    for (int j = 0; j < 32; ++j) a += wr[j] * mr[j];
    vh1F[(e * 3 + c) * 32 + h] = a;
  }
  __syncthreads();
  for (int i = tid; i < 16 * 32; i += 256) {  // ||vh1|| -> A1[256..287]
    int e = i / 32, h = i % 32;
    float x = vh1F[(e * 3 + 0) * 32 + h], y = vh1F[(e * 3 + 1) * 32 + h],
          z = vh1F[(e * 3 + 2) * 32 + h];
    A1[e * 288 + 256 + h] = (_Float16)sqrtf(fmaxf(x * x + y * y + z * z, 1e-8f));
  }
  __syncthreads();
  // msg1 s_out GEMM: [16,288] x [288,256] (act=False: no relu, gate_in = s)
  {
    const int n0 = wave * 32;
    v8f ac0 = wmma_gemm(A1, 288, B1, 256, n0, 9);
    v8f ac1 = wmma_gemm(A1, 288, B1, 256, n0 + 16, 9);
    const int nn = lane & 15, mB = (lane < 16) ? 0 : 8;
    for (int r = 0; r < 8; ++r) {
      int m = mB + r;
      float x0 = ac0[r] + wsb1[n0 + nn];
      so1[m * 256 + n0 + nn] = x0;
      GA[m * 256 + n0 + nn] = (_Float16)x0;
      float x1 = ac1[r] + wsb1[n0 + 16 + nn];
      so1[m * 256 + n0 + 16 + nn] = x1;
      GA[m * 256 + n0 + 16 + nn] = (_Float16)x1;
    }
  }
  __syncthreads();
  if (wave < 2) {  // gate1 GEMM
    const int n0 = wave * 16;
    v8f g = wmma_gemm(GA, 256, Bg1, 32, n0, 8);
    const int nn = lane & 15, mB = (lane < 16) ? 0 : 8;
    for (int r = 0; r < 8; ++r)
      g1F[(mB + r) * 32 + n0 + nn] = sigm(g[r] + wsvb1[n0 + nn]);
  }
  for (int i = tid; i < 16 * 96; i += 256) {  // v_out1 pre-gate (wv1 @ vh1)
    int e = i / 96, r = i % 96, c = r / 32, o = r % 32;
    const float* wr = wv1 + o * 32;
    const float* vr = vh1F + (e * 3 + c) * 32;
    float a = 0.f;
    for (int j = 0; j < 32; ++j) a += wr[j] * vr[j];
    vtF[(e * 3 + c) * 32 + o] = a;
  }
  __syncthreads();
  // scatter-add to destination nodes
  for (int i = tid; i < 16 * 256; i += 256) {
    int e = i >> 8, k = i & 255;
    atomicAdd(&ACC_S[dstI[e] * SH + k], so1[e * 256 + k]);
  }
  for (int i = tid; i < 16 * 96; i += 256) {
    int e = i / 96, r = i % 96, ch = r / 3, c = r % 3;
    float val = vtF[(e * 3 + c) * 32 + ch] * g1F[e * 32 + ch];
    atomicAdd(&ACC_V[dstI[e] * 96 + ch * 3 + c], val);
  }
}

// ------------------- fused node update + feed-forward (WMMA) ---------------
// one block = 16 nodes: seg-mean residual, gvp_ln0, ff0(K=320->1024),
// gates, ff1(K=1088->256), residual, gvp_ln1.
__global__ __launch_bounds__(256) void k_node(
    const float* DEG, float* S, float* Vv, const float* ACC_S,
    const float* ACC_V, const float* ln0g, const float* ln0b,
    const float* ln1g, const float* ln1b, const float* wh0f,
    const float* wsb0f, const float* wv0f, const float* wsvb0f,
    const float* wh1f, const float* wsb1f, const float* wv1f,
    const float* wsvb1f, const _Float16* Bf0, const _Float16* Bgf0,
    const _Float16* Bf1, const _Float16* Bgf1) {
  extern __shared__ char smem_raw[];
  _Float16* ffA = (_Float16*)smem_raw;  // 16x320
  _Float16* fA1 = ffA + 16 * 320;       // 16x1088
  _Float16* gA = fA1 + 16 * 1088;       // 16x1024 (reused 16x256 for ff1 gate)
  float* s1A = (float*)(gA + 16 * 1024);  // 16x256
  float* vA = s1A + 4096;                 // 16x3x32
  float* vh0F = vA + 1536;                // 16x3x64
  float* fvF = vh0F + 3072;               // 16x3x64
  float* vh1F = fvF + 3072;               // 16x3x64
  float* g0F = vh1F + 3072;               // 16x64
  float* g1F = g0F + 1024;                // 16x32
  float* stat = g1F + 512;                // mu[16], rs[16], vd[16]

  const int tid = threadIdx.x, wave = tid >> 5, lane = tid & 31;
  const int n0b = blockIdx.x * 16;

  // t_s = s + acc_s/deg ; t_v = v + acc_v/deg
  for (int i = tid; i < 16 * 256; i += 256) {
    int e = i >> 8, k = i & 255, n = n0b + e;
    float dg = fmaxf(DEG[n], 1.f);
    s1A[i] = S[n * SH + k] + ACC_S[n * SH + k] / dg;
  }
  for (int i = tid; i < 16 * 96; i += 256) {
    int e = i / 96, q = i % 96, ch = q / 3, c = q % 3, n = n0b + e;
    float dg = fmaxf(DEG[n], 1.f);
    vA[(e * 3 + c) * 32 + ch] =
        Vv[n * 96 + ch * 3 + c] + ACC_V[n * 96 + ch * 3 + c] / dg;
  }
  __syncthreads();
  if (tid < 16) {  // gvp_ln0 statistics
    int e = tid;
    float mu = 0.f;
    for (int j = 0; j < 256; ++j) mu += s1A[e * 256 + j];
    mu *= (1.f / 256.f);
    float var = 0.f;
    for (int j = 0; j < 256; ++j) { float d = s1A[e * 256 + j] - mu; var += d * d; }
    var *= (1.f / 256.f);
    stat[e] = mu;
    stat[16 + e] = rsqrtf(var + 1e-5f);
    float vm = 0.f;
    for (int ch = 0; ch < 32; ++ch) {
      float x = vA[(e * 3 + 0) * 32 + ch], y = vA[(e * 3 + 1) * 32 + ch],
            z = vA[(e * 3 + 2) * 32 + ch];
      vm += fmaxf(x * x + y * y + z * z, 1e-8f);
    }
    stat[32 + e] = rsqrtf(vm * (1.f / 32.f));
  }
  __syncthreads();
  for (int i = tid; i < 16 * 256; i += 256) {  // s1 = LN(t_s)
    int e = i >> 8, k = i & 255;
    float x = (s1A[i] - stat[e]) * stat[16 + e] * ln0g[k] + ln0b[k];
    s1A[i] = x;
    ffA[e * 320 + k] = (_Float16)x;
  }
  for (int i = tid; i < 16 * 96; i += 256) vA[i] *= stat[32 + (i / 96)];  // v1
  __syncthreads();
  for (int i = tid; i < 16 * 192; i += 256) {  // vh0 = wh0 @ v1  (64x32)
    int e = i / 192, r = i % 192, c = r / 64, h = r % 64;
    const float* wr = wh0f + h * 32;
    const float* vr = vA + (e * 3 + c) * 32;
    float a = 0.f;
    for (int j = 0; j < 32; ++j) a += wr[j] * vr[j];
    vh0F[(e * 3 + c) * 64 + h] = a;
  }
  __syncthreads();
  for (int i = tid; i < 16 * 64; i += 256) {  // ||vh0|| -> ffA[256..319]
    int e = i / 64, h = i % 64;
    float x = vh0F[(e * 3 + 0) * 64 + h], y = vh0F[(e * 3 + 1) * 64 + h],
          z = vh0F[(e * 3 + 2) * 64 + h];
    ffA[e * 320 + 256 + h] = (_Float16)sqrtf(fmaxf(x * x + y * y + z * z, 1e-8f));
  }
  __syncthreads();
  // ff0 s_out GEMM: [16,320] x [320,1024]; each wave does 8 N-tiles
  for (int nt = 0; nt < 8; ++nt) {
    const int n0 = (wave * 8 + nt) * 16;
    v8f ac = wmma_gemm(ffA, 320, Bf0, 1024, n0, 10);
    const int nn = lane & 15, mB = (lane < 16) ? 0 : 8;
    for (int r = 0; r < 8; ++r) {
      int m = mB + r;
      float x = ac[r] + wsb0f[n0 + nn];
      gA[m * 1024 + n0 + nn] = (_Float16)sigm(x);        // gate input
      fA1[m * 1088 + n0 + nn] = (_Float16)fmaxf(x, 0.f); // relu -> ff1 input
    }
  }
  __syncthreads();
  if (wave < 4) {  // ff0 gate GEMM [16,1024] x [1024,64]
    const int n0 = wave * 16;
    v8f g = wmma_gemm(gA, 1024, Bgf0, 64, n0, 32);
    const int nn = lane & 15, mB = (lane < 16) ? 0 : 8;
    for (int r = 0; r < 8; ++r)
      g0F[(mB + r) * 64 + n0 + nn] = sigm(g[r] + wsvb0f[n0 + nn]);
  }
  for (int i = tid; i < 16 * 192; i += 256) {  // wv0 @ vh0 (pre-gate)
    int e = i / 192, r = i % 192, c = r / 64, o = r % 64;
    const float* wr = wv0f + o * 64;
    const float* vr = vh0F + (e * 3 + c) * 64;
    float a = 0.f;
    for (int j = 0; j < 64; ++j) a += wr[j] * vr[j];
    fvF[(e * 3 + c) * 64 + o] = a;
  }
  __syncthreads();
  for (int i = tid; i < 16 * 192; i += 256)  // fv *= gate0
    fvF[i] *= g0F[(i / 192) * 64 + (i % 64)];
  __syncthreads();
  for (int i = tid; i < 16 * 192; i += 256) {  // vh1 = wh1 @ fv (64x64)
    int e = i / 192, r = i % 192, c = r / 64, h = r % 64;
    const float* wr = wh1f + h * 64;
    const float* vr = fvF + (e * 3 + c) * 64;
    float a = 0.f;
    for (int j = 0; j < 64; ++j) a += wr[j] * vr[j];
    vh1F[(e * 3 + c) * 64 + h] = a;
  }
  __syncthreads();
  for (int i = tid; i < 16 * 64; i += 256) {  // ||vh1|| -> fA1[1024..1087]
    int e = i / 64, h = i % 64;
    float x = vh1F[(e * 3 + 0) * 64 + h], y = vh1F[(e * 3 + 1) * 64 + h],
          z = vh1F[(e * 3 + 2) * 64 + h];
    fA1[e * 1088 + 1024 + h] = (_Float16)sqrtf(fmaxf(x * x + y * y + z * z, 1e-8f));
  }
  __syncthreads();
  // ff1 s_out GEMM: [16,1088] x [1088,256]; residual add into s1A
  for (int nt = 0; nt < 2; ++nt) {
    const int n0 = (wave * 2 + nt) * 16;
    v8f ac = wmma_gemm(fA1, 1088, Bf1, 256, n0, 34);
    const int nn = lane & 15, mB = (lane < 16) ? 0 : 8;
    for (int r = 0; r < 8; ++r) {
      int m = mB + r;
      float x = ac[r] + wsb1f[n0 + nn];
      s1A[m * 256 + n0 + nn] += x;          // u_s = s1 + fs
      gA[m * 256 + n0 + nn] = (_Float16)x;  // gate_in (act=False)
    }
  }
  __syncthreads();
  if (wave < 2) {  // ff1 gate GEMM [16,256] x [256,32]
    const int n0 = wave * 16;
    v8f g = wmma_gemm(gA, 256, Bgf1, 32, n0, 8);
    const int nn = lane & 15, mB = (lane < 16) ? 0 : 8;
    for (int r = 0; r < 8; ++r)
      g1F[(mB + r) * 32 + n0 + nn] = sigm(g[r] + wsvb1f[n0 + nn]);
  }
  __syncthreads();
  for (int i = tid; i < 16 * 96; i += 256) {  // u_v = v1 + wv1@vh1 * gate1
    int e = i / 96, q = i % 96, c = q / 32, o = q % 32;
    const float* wr = wv1f + o * 64;
    const float* vr = vh1F + (e * 3 + c) * 64;
    float a = 0.f;
    for (int j = 0; j < 64; ++j) a += wr[j] * vr[j];
    vA[(e * 3 + c) * 32 + o] += a * g1F[e * 32 + o];
  }
  __syncthreads();
  if (tid < 16) {  // gvp_ln1 statistics
    int e = tid;
    float mu = 0.f;
    for (int j = 0; j < 256; ++j) mu += s1A[e * 256 + j];
    mu *= (1.f / 256.f);
    float var = 0.f;
    for (int j = 0; j < 256; ++j) { float d = s1A[e * 256 + j] - mu; var += d * d; }
    var *= (1.f / 256.f);
    stat[e] = mu;
    stat[16 + e] = rsqrtf(var + 1e-5f);
    float vm = 0.f;
    for (int ch = 0; ch < 32; ++ch) {
      float x = vA[(e * 3 + 0) * 32 + ch], y = vA[(e * 3 + 1) * 32 + ch],
            z = vA[(e * 3 + 2) * 32 + ch];
      vm += fmaxf(x * x + y * y + z * z, 1e-8f);
    }
    stat[32 + e] = rsqrtf(vm * (1.f / 32.f));
  }
  __syncthreads();
  for (int i = tid; i < 16 * 256; i += 256) {
    int e = i >> 8, k = i & 255, n = n0b + e;
    S[n * SH + k] = (s1A[i] - stat[e]) * stat[16 + e] * ln1g[k] + ln1b[k];
  }
  for (int i = tid; i < 16 * 96; i += 256) {
    int e = i / 96, q = i % 96, ch = q / 3, c = q % 3, n = n0b + e;
    Vv[n * 96 + ch * 3 + c] = vA[(e * 3 + c) * 32 + ch] * stat[32 + e];
  }
}

// ----------------------------- edge feature update -------------------------
__global__ __launch_bounds__(256) void k_edge(
    const int* EI, const float* S, float* ES, const float* DIST,
    const float* w1, const float* b1, const float* w2, const float* b2,
    const float* lg, const float* lb) {
  __shared__ float fL[16 * 544];
  __shared__ float hL[256];
  __shared__ float tL[256];
  __shared__ int sI[16], dI[16];
  const int tid = threadIdx.x;
  const int e0 = blockIdx.x * 16;
  if (tid < 16) { sI[tid] = EI[e0 + tid]; dI[tid] = EI[NE + e0 + tid]; }
  __syncthreads();
  for (int i = tid; i < 16 * 544; i += 256) {
    int e = i / 544, k = i % 544;
    float v;
    if (k < 256) v = S[sI[e] * SH + k];
    else if (k < 512) v = S[dI[e] * SH + (k - 256)];
    else if (k < 528) v = ES[(e0 + e) * 16 + (k - 512)];
    else v = DIST[(e0 + e) * 16 + (k - 528)];
    fL[i] = v;
  }
  __syncthreads();
  {
    int e = tid >> 4, o = tid & 15;
    float a = b1[o];
    const float* wr = w1 + o * 544;
    const float* fr = fL + e * 544;
    for (int k = 0; k < 544; ++k) a += wr[k] * fr[k];
    hL[e * 16 + o] = fmaxf(a, 0.f);
  }
  __syncthreads();
  {
    int e = tid >> 4, o = tid & 15;
    float a = b2[o];
    for (int j = 0; j < 16; ++j) a += w2[o * 16 + j] * hL[e * 16 + j];
    tL[e * 16 + o] = ES[(e0 + e) * 16 + o] + a;
  }
  __syncthreads();
  {
    int e = tid >> 4, o = tid & 15;
    float mu = 0.f;
    for (int j = 0; j < 16; ++j) mu += tL[e * 16 + j];
    mu *= (1.f / 16.f);
    float var = 0.f;
    for (int j = 0; j < 16; ++j) { float d = tL[e * 16 + j] - mu; var += d * d; }
    var *= (1.f / 16.f);
    ES[(e0 + e) * 16 + o] =
        (tL[e * 16 + o] - mu) * rsqrtf(var + 1e-5f) * lg[o] + lb[o];
  }
}

// --------------------------------- readout ---------------------------------
__global__ __launch_bounds__(256) void k_readout(
    const float* S, const float* Vv, const int* RIDX, const float* w1,
    const float* b1, const float* lg, const float* lb, const float* w2,
    const float* b2, float* OUT) {
  __shared__ float dn[2 * 288];
  __shared__ float aL[2 * 128];
  __shared__ float arL[2 * 128];
  const int tid = threadIdx.x;
  const int n0 = blockIdx.x * 2;
  for (int i = tid; i < 2 * 288; i += 256) {
    int e = i / 288, k = i % 288, n = n0 + e;
    float v;
    if (k < 256) v = S[n * SH + k];
    else {
      int ch = k - 256;
      float x = Vv[n * 96 + ch * 3 + 0], y = Vv[n * 96 + ch * 3 + 1],
            z = Vv[n * 96 + ch * 3 + 2];
      v = sqrtf(fmaxf(x * x + y * y + z * z, 1e-8f));
    }
    dn[i] = v;
  }
  __syncthreads();
  {
    int e = tid >> 7, o = tid & 127;
    float a = b1[o];
    const float* wr = w1 + o * 288;
    const float* dr = dn + e * 288;
    for (int k = 0; k < 288; ++k) a += wr[k] * dr[k];
    aL[e * 128 + o] = a;
  }
  __syncthreads();
  {
    int e = tid >> 7, o = tid & 127;
    float mu = 0.f;
    for (int j = 0; j < 128; ++j) mu += aL[e * 128 + j];
    mu *= (1.f / 128.f);
    float var = 0.f;
    for (int j = 0; j < 128; ++j) { float d = aL[e * 128 + j] - mu; var += d * d; }
    var *= (1.f / 128.f);
    arL[e * 128 + o] =
        fmaxf((aL[e * 128 + o] - mu) * rsqrtf(var + 1e-5f) * lg[o] + lb[o], 0.f);
  }
  __syncthreads();
  {
    int e = tid >> 7, o = tid & 127, n = n0 + e;
    float a = b2[o];
    for (int j = 0; j < 128; ++j) a += w2[o * 128 + j] * arL[e * 128 + j];
    atomicAdd(&OUT[RIDX[n] * 128 + o], a);
  }
}

// ---------------------------------------------------------------------------
extern "C" void kernel_launch(void* const* d_in, const int* in_sizes, int n_in,
                              void* d_out, int out_size, void* d_ws,
                              size_t ws_size, hipStream_t stream) {
  (void)in_sizes; (void)n_in; (void)out_size; (void)ws_size;
  auto P = [&](int i) { return (const float*)d_in[i]; };
  const float* X = P(0);
  const float* POS = P(1);
  const int* EI = (const int*)d_in[2];
  const int* RIDX = (const int*)d_in[3];
  float* OUT = (float*)d_out;

  // ---- workspace carve (floats, 64-elt aligned) ----
  float* wsF = (float*)d_ws;
  size_t off = 0;
  auto carve = [&](size_t n) -> float* {
    float* p = wsF + off;
    off += (n + 63) & ~(size_t)63;
    return p;
  };
  float* S = carve((size_t)NN * SH);
  float* Vv = carve((size_t)NN * 96);
  float* ES = carve((size_t)NE * 16);
  float* DIST = carve((size_t)NE * 16);
  float* U = carve((size_t)NE * 3);
  float* ACC_S = carve((size_t)NN * SH);
  float* ACC_V = carve((size_t)NN * 96);
  float* DEG = carve(NN);
  float* RCNT = carve(NRES);
  const size_t WSTRIDE = 925696;  // f16 elements per layer
  _Float16* W16 = (_Float16*)carve((3 * WSTRIDE + 1) / 2);

  // ---- zero accumulators / counters / output ----
  k_zero<<<(NN + 255) / 256, 256, 0, stream>>>(DEG, NN);
  k_zero<<<(NRES + 255) / 256, 256, 0, stream>>>(RCNT, NRES);
  k_zero<<<(NRES * 128 + 255) / 256, 256, 0, stream>>>(OUT, NRES * 128);

  // ---- weight prep: f32 [O][K] -> f16 [Kpad][O] ----
  auto prep = [&](const float* W, _Float16* Bt, int O, int K, int Kpad) {
    int tot = Kpad * O;
    k_prep<<<(tot + 255) / 256, 256, 0, stream>>>(W, Bt, O, K, Kpad);
  };
  for (int l = 0; l < 3; ++l) {
    const int Lb = 17 + l * 28;
    _Float16* Wl = W16 + (size_t)l * WSTRIDE;
    prep(P(Lb + 1), Wl + 0, 256, 593, 608);        // msg0.ws_w
    prep(P(Lb + 4), Wl + 155648, 32, 256, 256);    // msg0.wsv_w
    prep(P(Lb + 7), Wl + 163840, 256, 288, 288);   // msg1.ws_w
    prep(P(Lb + 10), Wl + 237568, 32, 256, 256);   // msg1.wsv_w
    prep(P(Lb + 15), Wl + 245760, 1024, 320, 320); // ff0.ws_w
    prep(P(Lb + 18), Wl + 573440, 64, 1024, 1024); // ff0.wsv_w
    prep(P(Lb + 21), Wl + 638976, 256, 1088, 1088);// ff1.ws_w
    prep(P(Lb + 24), Wl + 917504, 32, 256, 256);   // ff1.wsv_w
  }

  // ---- encoder + input GVP, geometry, degree counts ----
  k_encode<<<NN, 64, 0, stream>>>(X, P(5), P(6), P(7), P(8), P(9), P(10),
                                  P(12), P(13), S, Vv);
  k_geom<<<(NE + 255) / 256, 256, 0, stream>>>(EI, POS, U, DIST, ES);
  k_deg<<<(NE + 255) / 256, 256, 0, stream>>>(EI, DEG);
  k_rcnt<<<(NN + 255) / 256, 256, 0, stream>>>(RIDX, RCNT);

  constexpr int MSG_SMEM = (16 * 608 + 16 * 288 + 16 * 256) * 2 +
                           (3120 + 3120 + 1536 + 1536 + 1536 + 512 + 512 + 4096) * 4 +
                           32 * 4;
  constexpr int NODE_SMEM = (16 * 320 + 16 * 1088 + 16 * 1024) * 2 +
                            (4096 + 1536 + 3072 + 3072 + 3072 + 1024 + 512 + 48) * 4;

  for (int l = 0; l < 3; ++l) {
    const int Lb = 17 + l * 28;
    _Float16* Wl = W16 + (size_t)l * WSTRIDE;
    k_zero<<<(NN * SH + 255) / 256, 256, 0, stream>>>(ACC_S, NN * SH);
    k_zero<<<(NN * 96 + 255) / 256, 256, 0, stream>>>(ACC_V, NN * 96);
    k_msg<<<NE / 16, 256, MSG_SMEM, stream>>>(
        EI, S, Vv, ES, U,
        P(Lb + 0), P(Lb + 2), P(Lb + 3), P(Lb + 5),   // msg0 wh, ws_b, wv, wsv_b
        P(Lb + 6), P(Lb + 8), P(Lb + 9), P(Lb + 11),  // msg1
        Wl + 0, Wl + 155648, Wl + 163840, Wl + 237568, ACC_S, ACC_V);
    k_node<<<NN / 16, 256, NODE_SMEM, stream>>>(
        DEG, S, Vv, ACC_S, ACC_V, P(Lb + 12), P(Lb + 13), P(Lb + 26), P(Lb + 27),
        P(Lb + 14), P(Lb + 16), P(Lb + 17), P(Lb + 19),  // ff0 wh, ws_b, wv, wsv_b
        P(Lb + 20), P(Lb + 22), P(Lb + 23), P(Lb + 25),  // ff1
        Wl + 245760, Wl + 573440, Wl + 638976, Wl + 917504);
    if (l + 1 < 3) {  // es never read again after the last layer
      k_edge<<<NE / 16, 256, 0, stream>>>(EI, S, ES, DIST, P(101), P(102),
                                          P(103), P(104), P(105), P(106));
    }
  }

  k_readout<<<NN / 2, 256, 0, stream>>>(S, Vv, RIDX, P(107), P(108), P(109),
                                        P(110), P(111), P(112), OUT);
  k_div<<<(NRES * 128 + 255) / 256, 256, 0, stream>>>(OUT, RCNT);
}
